// SelfAttention_65566970741004
// MI455X (gfx1250) — matrix operand order repaired and checked by reading, not compiled, not run on hardware
//
#include <hip/hip_runtime.h>
#include <stdint.h>

typedef __attribute__((ext_vector_type(16))) _Float16 v16h;
typedef __attribute__((ext_vector_type(8)))  _Float16 v8h;
typedef __attribute__((ext_vector_type(8)))  float    v8f;

#define CC   256
#define TT   4096
#define BB   8
#define GG   32
#define GC   8          // channels per group
#define EPSV 1e-5f

// ---------------------------------------------------------------------------
// WMMA helpers (CDNA5: D = A(16x32 f16) * B(32x16 f16) + C(16x16 f32))
// ---------------------------------------------------------------------------
__device__ __forceinline__ v8f wmma_f16(v16h a, v16h b, v8f c) {
  return __builtin_amdgcn_wmma_f32_16x16x32_f16(false, a, false, b,
                                                (short)0, c, false, false);
}

// A fragment: source row-major (M x K), row stride ldk (halfs).
// lanes 0-15: m=lane, k = k0..k0+15
// lanes16-31: m=lane-16, k = k0+16..k0+31
__device__ __forceinline__ v16h load_a_frag(const _Float16* A, int ldk,
                                            int lane, int k0) {
  int m  = lane & 15;
  int kA = k0 + ((lane & 16) ? 16 : 0);
  const v8h* p0 = (const v8h*)(A + (size_t)m * ldk + kA);
  const v8h* p1 = (const v8h*)(A + (size_t)m * ldk + kA + 8);
  v8h lo = *p0, hi = *p1;
  v16h r;
#pragma unroll
  for (int i = 0; i < 8; ++i) { r[i] = lo[i]; r[i + 8] = hi[i]; }
  return r;
}

// B fragment: source is B^T, i.e. row-major (N x K), row stride ldk (halfs).
// lanes 0-15: n=lane,   k = k0..k0+15 ; lanes16-31: n=lane-16, k = k0+16..k0+31
__device__ __forceinline__ v16h load_b_frag(const _Float16* Bt, int ldk,
                                            int lane, int k0) {
  int n = lane & 15;
  int k = k0 + ((lane & 16) ? 16 : 0);
  return *(const v16h*)(Bt + (size_t)n * ldk + k);
}

// CDNA5 async global->LDS copy: 16 bytes per lane, ASYNCcnt-tracked,
// bypasses VGPRs (CDNA5 ISA).
__device__ __forceinline__ void async_copy_b128(const _Float16* gptr,
                                                _Float16* lptr) {
  unsigned la = (unsigned)(uintptr_t)lptr;          // low 32 bits = LDS offset
  unsigned long long ga = (unsigned long long)gptr;
  asm volatile("global_load_async_to_lds_b128 %0, %1, off"
               :: "v"(la), "v"(ga) : "memory");
}
__device__ __forceinline__ void async_wait0() {
  asm volatile("s_wait_asynccnt 0x0" ::: "memory");
}

// ---------------------------------------------------------------------------
// 1) GroupNorm statistics: one block per (b,g); 8 ch x 4096 px reduction
// ---------------------------------------------------------------------------
__global__ void gn_stats_kernel(const float* __restrict__ x,
                                float* __restrict__ stats) {
  int bg = blockIdx.x;                 // b*32+g
  const float* p = x + (size_t)bg * GC * TT;
  float s = 0.f, ss = 0.f;
  for (int i = threadIdx.x; i < GC * TT; i += 256) {
    float v = p[i]; s += v; ss += v * v;
  }
  __shared__ float red[512];
  red[threadIdx.x] = s; red[256 + threadIdx.x] = ss;
  __syncthreads();
  for (int st = 128; st > 0; st >>= 1) {
    if (threadIdx.x < st) {
      red[threadIdx.x]       += red[threadIdx.x + st];
      red[256 + threadIdx.x] += red[256 + threadIdx.x + st];
    }
    __syncthreads();
  }
  if (threadIdx.x == 0) {
    float inv = 1.f / (GC * TT);
    float mu  = red[0] * inv;
    float var = red[256] * inv - mu * mu;
    stats[bg * 2]     = mu;
    stats[bg * 2 + 1] = rsqrtf(var + EPSV);
  }
}

// ---------------------------------------------------------------------------
// 2) GroupNorm apply + (C,T)->(T,C) transpose, f32 -> f16
// grid (T/64, C/32, B), 256 threads
// ---------------------------------------------------------------------------
__global__ void gn_apply_kernel(const float* __restrict__ x,
                                const float* __restrict__ gamma,
                                const float* __restrict__ beta,
                                const float* __restrict__ stats,
                                _Float16* __restrict__ H) {
  __shared__ _Float16 tile[32][72];
  int b = blockIdx.z, c0 = blockIdx.y * 32, t0 = blockIdx.x * 64;
  int tid = threadIdx.x;
  int tl = tid & 63, cq = tid >> 6;
#pragma unroll
  for (int r = 0; r < 8; ++r) {
    int cl = r * 4 + cq;
    int c  = c0 + cl;
    float mu = stats[(b * GG + (c >> 3)) * 2];
    float rs = stats[(b * GG + (c >> 3)) * 2 + 1];
    float v  = x[(size_t)b * CC * TT + (size_t)c * TT + t0 + tl];
    tile[cl][tl] = (_Float16)(((v - mu) * rs) * gamma[c] + beta[c]);
  }
  __syncthreads();
  int cl2 = tid & 31, tq = tid >> 5;
#pragma unroll
  for (int w = 0; w < 8; ++w) {
    int tl2 = w * 8 + tq;
    H[(size_t)b * TT * CC + (size_t)(t0 + tl2) * CC + c0 + cl2] = tile[cl2][tl2];
  }
}

// ---------------------------------------------------------------------------
// 3) Weight convert f32 -> f16 (wq, wk, wv, wp laid out contiguously)
// ---------------------------------------------------------------------------
__global__ void wcvt_kernel(const float* __restrict__ wq,
                            const float* __restrict__ wk,
                            const float* __restrict__ wv,
                            const float* __restrict__ wp,
                            _Float16* __restrict__ out) {
  int i = blockIdx.x * 256 + threadIdx.x;       // 0 .. 4*65536-1
  const float* srcs[4] = {wq, wk, wv, wp};
  out[i] = (_Float16)srcs[i >> 16][i & 65535];
}

// ---------------------------------------------------------------------------
// 4) Generic WMMA GEMM: one wave per 16(M) x 64(N) output strip.
//    D[m][n] = sum_k A[m][k] * Bt[n][k];  out = (D + bias) * alpha (+resid)
//    A fragment is reused across 4 N-tiles (10 b128 loads per 4 WMMAs).
// grid (M/16, N/64, B), block 32
// ---------------------------------------------------------------------------
__global__ void gemm16x64_kernel(const _Float16* __restrict__ A, size_t aBatch, int lda,
                                 const _Float16* __restrict__ Bt, size_t bBatch, int ldb,
                                 const float* __restrict__ bias, int biasPerM,
                                 float alpha,
                                 _Float16* __restrict__ outH,
                                 float* __restrict__ outF,
                                 const float* __restrict__ resid,
                                 size_t oBatch, int ldo) {
  int lane = threadIdx.x;
  int m0 = blockIdx.x * 16, n0 = blockIdx.y * 64, b = blockIdx.z;
  const _Float16* Ab = A  + aBatch * b + (size_t)m0 * lda;
  const _Float16* Bb = Bt + bBatch * b + (size_t)n0 * ldb;
  v8f zero = {};
  v8f acc[4];
#pragma unroll
  for (int j = 0; j < 4; ++j) acc[j] = zero;
#pragma unroll
  for (int k0 = 0; k0 < CC; k0 += 32) {
    v16h af = load_a_frag(Ab, lda, lane, k0);
#pragma unroll
    for (int j = 0; j < 4; ++j) {
      v16h bf = load_b_frag(Bb + (size_t)j * 16 * ldb, ldb, lane, k0);
      acc[j] = wmma_f16(af, bf, acc[j]);
    }
  }
  int mbase = m0 + ((lane & 16) ? 8 : 0);
#pragma unroll
  for (int j = 0; j < 4; ++j) {
    int n = n0 + j * 16 + (lane & 15);
#pragma unroll
    for (int r = 0; r < 8; ++r) {
      int m = mbase + r;
      float bv = bias ? (biasPerM ? bias[m] : bias[n]) : 0.f;
      float val = (acc[j][r] + bv) * alpha;
      size_t oi = oBatch * b + (size_t)m * ldo + n;
      if (outF) outF[oi] = resid[oi] + val;
      else      outH[oi] = (_Float16)val;
    }
  }
}

// ---------------------------------------------------------------------------
// 5) Flash attention: grid (T/128, B), 256 threads = 8 waves.
//    Wave w owns 16 queries; loops 32-key tiles staged in LDS via
//    GLOBAL_LOAD_ASYNC_TO_LDS_B128 (ASYNCcnt) with online softmax;
//    O accumulated in 16 x v8f registers.
//    Qm,Km: (T,C) f16; Vm: (C,T) f16; Z: (T,C) f16.
// ---------------------------------------------------------------------------
__global__ void __launch_bounds__(256, 1)
attn_kernel(const _Float16* __restrict__ Qm,
            const _Float16* __restrict__ Km,
            const _Float16* __restrict__ Vm,
            _Float16* __restrict__ Z) {
  __shared__ _Float16 sK[32][272];      // 32 keys x 256 ch (+pad)
  __shared__ _Float16 sV[256][48];      // 256 ch  x 32 keys (+pad)
  __shared__ _Float16 sP[8][16][48];    // per-wave P scratch 16x32 (+pad)

  int b    = blockIdx.y;
  int tid  = threadIdx.x;
  int wave = tid >> 5, lane = tid & 31;
  int q0   = blockIdx.x * 128 + wave * 16;

  const _Float16* Qb = Qm + (size_t)b * TT * CC + (size_t)q0 * CC;
  const _Float16* Kb = Km + (size_t)b * TT * CC;
  const _Float16* Vb = Vm + (size_t)b * CC * TT;

  // resident Q fragments (16 queries x 256 channels)
  v16h qa[8];
#pragma unroll
  for (int kk = 0; kk < 8; ++kk) qa[kk] = load_a_frag(Qb, CC, lane, kk * 32);

  v8f zero = {};
  v8f o[16];
#pragma unroll
  for (int n = 0; n < 16; ++n) o[n] = zero;
  float mrow[8], lsum[8];
#pragma unroll
  for (int r = 0; r < 8; ++r) { mrow[r] = -1e30f; lsum[r] = 0.f; }

  for (int s0 = 0; s0 < TT; s0 += 32) {
    // --- async-stage K tile (32 x 256) and V^T tile (256 x 32) into LDS ---
    // Each thread issues 4+4 16-byte async transfers (8-half chunks).
#pragma unroll
    for (int i = 0; i < 4; ++i) {
      int ch   = tid * 4 + i;                 // 0..1023 chunks of 8 halfs
      int row  = ch >> 5, col8 = (ch & 31) * 8;
      async_copy_b128(Kb + (size_t)(s0 + row) * CC + col8, &sK[row][col8]);
    }
#pragma unroll
    for (int i = 0; i < 4; ++i) {
      int ch   = tid * 4 + i;
      int row  = ch >> 2, col8 = (ch & 3) * 8;
      async_copy_b128(Vb + (size_t)row * TT + s0 + col8, &sV[row][col8]);
    }
    async_wait0();
    __syncthreads();

    // --- scores: S(16x32) = Q(16x256) * K_tile^T ---
    v8f sc0 = zero, sc1 = zero;
#pragma unroll
    for (int kk = 0; kk < 8; ++kk) {
      int k = kk * 32 + ((lane & 16) ? 16 : 0);
      v16h b0 = *(const v16h*)&sK[0  + (lane & 15)][k];
      v16h b1 = *(const v16h*)&sK[16 + (lane & 15)][k];
      sc0 = wmma_f16(qa[kk], b0, sc0);
      sc1 = wmma_f16(qa[kk], b1, sc1);
    }

    // --- online softmax over the 32 new columns ---
    float scalef[8];
#pragma unroll
    for (int r = 0; r < 8; ++r) {
      float v = fmaxf(sc0[r], sc1[r]);
      v = fmaxf(v, __shfl_xor(v, 1));
      v = fmaxf(v, __shfl_xor(v, 2));
      v = fmaxf(v, __shfl_xor(v, 4));
      v = fmaxf(v, __shfl_xor(v, 8));
      float mnew = fmaxf(mrow[r], v);
      scalef[r]  = __expf(mrow[r] - mnew);
      mrow[r]    = mnew;
      float p0 = __expf(sc0[r] - mnew);
      float p1 = __expf(sc1[r] - mnew);
      sc0[r] = p0; sc1[r] = p1;
      float rs = p0 + p1;
      rs += __shfl_xor(rs, 1);
      rs += __shfl_xor(rs, 2);
      rs += __shfl_xor(rs, 4);
      rs += __shfl_xor(rs, 8);
      lsum[r] = lsum[r] * scalef[r] + rs;
    }
#pragma unroll
    for (int n = 0; n < 16; ++n)
#pragma unroll
      for (int r = 0; r < 8; ++r) o[n][r] *= scalef[r];

    // --- P (D layout) -> LDS -> A-fragment layout ---
    {
      int mb = (lane & 16) ? 8 : 0, col = lane & 15;
#pragma unroll
      for (int r = 0; r < 8; ++r) {
        sP[wave][mb + r][col]      = (_Float16)sc0[r];
        sP[wave][mb + r][col + 16] = (_Float16)sc1[r];
      }
    }
    __syncthreads();

    // --- O += P(16x32) * V_tile(32x256) ---
    v16h pf = load_a_frag(&sP[wave][0][0], 48, lane, 0);
#pragma unroll
    for (int n = 0; n < 16; ++n) {
      int k = (lane & 16) ? 16 : 0;
      v16h vf = *(const v16h*)&sV[n * 16 + (lane & 15)][k];
      o[n] = wmma_f16(pf, vf, o[n]);
    }
    __syncthreads();
  }

  // --- normalize and write Z (T,C) f16 ---
  int mb = (lane & 16) ? 8 : 0;
#pragma unroll
  for (int r = 0; r < 8; ++r) {
    float inv = 1.f / lsum[r];
    int t = q0 + mb + r;
#pragma unroll
    for (int n = 0; n < 16; ++n) {
      int c = n * 16 + (lane & 15);
      Z[(size_t)b * TT * CC + (size_t)t * CC + c] = (_Float16)(o[n][r] * inv);
    }
  }
}

// ---------------------------------------------------------------------------
// Host-side launcher
// ---------------------------------------------------------------------------
extern "C" void kernel_launch(void* const* d_in, const int* in_sizes, int n_in,
                              void* d_out, int out_size, void* d_ws, size_t ws_size,
                              hipStream_t stream) {
  const float* x     = (const float*)d_in[0];
  const float* gamma = (const float*)d_in[1];
  const float* beta  = (const float*)d_in[2];
  const float* wq    = (const float*)d_in[3];
  const float* bq    = (const float*)d_in[4];
  const float* wk    = (const float*)d_in[5];
  const float* bk    = (const float*)d_in[6];
  const float* wv    = (const float*)d_in[7];
  const float* bv    = (const float*)d_in[8];
  const float* wp    = (const float*)d_in[9];
  const float* bp    = (const float*)d_in[10];

  size_t off = 0;
  char* wsb = (char*)d_ws;
  auto take = [&](size_t bytes) -> void* {
    void* p = wsb + off;
    off += (bytes + 255) & ~(size_t)255;
    return p;
  };
  const size_t fTC = (size_t)BB * TT * CC * sizeof(_Float16);
  float*    stats = (float*)take((size_t)BB * GG * 2 * sizeof(float));
  _Float16* w16   = (_Float16*)take((size_t)4 * CC * CC * sizeof(_Float16));
  _Float16* Hm    = (_Float16*)take(fTC);
  _Float16* Qm    = (_Float16*)take(fTC);
  _Float16* Km    = (_Float16*)take(fTC);
  _Float16* Vm    = (_Float16*)take(fTC);
  _Float16* Zm    = (_Float16*)take(fTC);
  _Float16* wq16 = w16;
  _Float16* wk16 = w16 + 1 * CC * CC;
  _Float16* wv16 = w16 + 2 * CC * CC;
  _Float16* wp16 = w16 + 3 * CC * CC;
  (void)in_sizes; (void)n_in; (void)out_size; (void)ws_size;

  // 1) GroupNorm stats + apply (writes H in (T,C) f16)
  gn_stats_kernel<<<BB * GG, 256, 0, stream>>>(x, stats);
  gn_apply_kernel<<<dim3(TT / 64, CC / 32, BB), 256, 0, stream>>>(
      x, gamma, beta, stats, Hm);

  // 2) Weights -> f16
  wcvt_kernel<<<(4 * CC * CC) / 256, 256, 0, stream>>>(wq, wk, wv, wp, w16);

  const size_t sTC = (size_t)TT * CC;

  // 3) Q = (H*WqT + bq) / sqrt(C)   -> (T,C)
  gemm16x64_kernel<<<dim3(TT / 16, CC / 64, BB), 32, 0, stream>>>(
      Hm, sTC, CC, wq16, 0, CC, bq, 0, 0.0625f,
      Qm, nullptr, nullptr, sTC, CC);
  //    K = H*WkT + bk               -> (T,C)
  gemm16x64_kernel<<<dim3(TT / 16, CC / 64, BB), 32, 0, stream>>>(
      Hm, sTC, CC, wk16, 0, CC, bk, 0, 1.0f,
      Km, nullptr, nullptr, sTC, CC);
  //    V = Wv*H^T + bv              -> (C,T)
  gemm16x64_kernel<<<dim3(CC / 16, TT / 64, BB), 32, 0, stream>>>(
      wv16, 0, CC, Hm, sTC, CC, bv, 1, 1.0f,
      Vm, nullptr, nullptr, sTC, TT);

  // 4) Flash attention -> Z (T,C)
  attn_kernel<<<dim3(TT / 128, BB), 256, 0, stream>>>(Qm, Km, Vm, Zm);

  // 5) out = x + Wp*Z^T + bp        -> (B,C,T) f32
  gemm16x64_kernel<<<dim3(CC / 16, TT / 64, BB), 32, 0, stream>>>(
      wp16, 0, CC, Zm, sTC, CC, bp, 1, 1.0f,
      nullptr, (float*)d_out, x, sTC, TT);
}
